// GCN_32822140076406
// MI455X (gfx1250) — matrix operand order repaired and checked by reading, not compiled, not run on hardware
//
#include <hip/hip_runtime.h>

typedef __attribute__((ext_vector_type(16))) __bf16 v16bf;
typedef __attribute__((ext_vector_type(8)))  float  v8f;

__device__ __forceinline__ unsigned short f32_to_bf16(float f) {
    union { float f; unsigned u; } v; v.f = f;
    unsigned r = v.u + 0x7FFFu + ((v.u >> 16) & 1u);   // round-to-nearest-even
    return (unsigned short)(r >> 16);
}

// Native CDNA5 global atomic fadd (no return -> STOREcnt path, no CAS loop)
__device__ __forceinline__ void atomic_add_f32(float* addr, float v) {
    asm volatile("global_atomic_add_f32 %0, %1, off" :: "v"(addr), "v"(v) : "memory");
}

// ---------------------------------------------------------------------------
// Propagate: out[dst] += w * h[src], F = 128. One wave per EPW=8 edges,
// 4 floats per lane per edge. Edge index/weight are wave-uniform: the wave id
// is pinned to an SGPR so metadata loads go through SMEM / constant cache.
// ---------------------------------------------------------------------------
__global__ __launch_bounds__(256)
void propagate_kernel(const float* __restrict__ h,
                      const long long* __restrict__ src,
                      const long long* __restrict__ dst,
                      const float* __restrict__ w,
                      float* __restrict__ out, int E) {
    constexpr int EPW = 8;                         // edges per wave
    const int lane = threadIdx.x & 31;
    int waveId = (int)(((long long)blockIdx.x * blockDim.x + threadIdx.x) >> 5);
    waveId = __builtin_amdgcn_readfirstlane(waveId);     // wave-uniform -> SGPR
    const int eBase = waveId * EPW;

    #pragma unroll
    for (int j = 0; j < EPW; ++j) {
        const int e = eBase + j;
        if (e >= E) break;                         // scalar branch
        const long long s  = src[e];               // s_load_b64
        const long long d  = dst[e];               // s_load_b64
        const float     we = w[e];                 // s_load_b32
        const float4 v = *(const float4*)(h + s * 128 + lane * 4);  // b128 gather
        float* o = out + d * 128 + lane * 4;
        atomic_add_f32(o + 0, v.x * we);
        atomic_add_f32(o + 1, v.y * we);
        atomic_add_f32(o + 2, v.z * we);
        atomic_add_f32(o + 3, v.w * we);
    }
}

// ---------------------------------------------------------------------------
// GEMM + bias (+ReLU): out[M x NOUT] = A[M x 128] * W[128 x NOUT], bf16 WMMA
// with f32 accumulate. 256 threads = 8 waves. LDS: W transposed (sW[n][k]) +
// A tile, both bf16, row stride 136 shorts (272B) -> 16B-aligned ds_load_b128
// fragments, conflict-free (dword stride 68 -> bank base 4*row).
// ---------------------------------------------------------------------------
template<int NOUT, int MLOOP, bool RELU>
__global__ __launch_bounds__(256)
void gemm_bias_act(const float* __restrict__ A, const float* __restrict__ W,
                   const float* __restrict__ bias, float* __restrict__ out, int M) {
    constexpr int K   = 128;
    constexpr int LDW = 136;                 // padded stride in shorts
    constexpr int NT  = NOUT / 16;           // N tiles
    constexpr int MW  = 8 / NT;              // waves along M
    constexpr int ROWS = 16 * MW * MLOOP;    // rows per block (= 64)

    __shared__ unsigned short sW[NOUT * LDW];
    __shared__ unsigned short sA[ROWS * LDW];

    const int tid = threadIdx.x;
    const int blockRow = blockIdx.x * ROWS;

    // Stage W transposed: sW[n*LDW + k] = bf16(W[k*NOUT + n])
    for (int i = tid; i < K * NOUT; i += 256) {
        int k = i / NOUT, n = i % NOUT;
        sW[n * LDW + k] = f32_to_bf16(W[i]);
    }
    // Stage A tile: sA[m*LDW + k]
    for (int i = tid; i < ROWS * K; i += 256) {
        int m = i >> 7, k = i & 127;
        int r = blockRow + m;
        sA[m * LDW + k] = (r < M) ? f32_to_bf16(A[(size_t)r * K + k])
                                  : (unsigned short)0;
    }
    __syncthreads();

    // All 256 threads active here: pin wave id to SGPR so tile indices and the
    // epilogue range check stay on the scalar path.
    const int wave  = __builtin_amdgcn_readfirstlane(tid >> 5);
    const int lane  = tid & 31;
    const int mWave = wave / NT;
    const int nTile = wave % NT;
    const int half  = lane >> 4;   // 0: lanes 0-15, 1: lanes 16-31
    const int l16   = lane & 15;

    // ISA 16-bit layouts: A lane<16 holds K 0..7 & 16..23, lane>=16 K 8..15 & 24..31
    //                     B lane<16 holds K 0..7 & 16..23, lane>=16 K 8..15 & 24..31
    const int aOff = half * 8;
    const int bOff = half * 8;
    const unsigned short* bRow = &sW[(nTile * 16 + l16) * LDW];

    const int   n  = nTile * 16 + l16;
    const float bn = bias[n];

    union Frag { v16bf v; uint4 q[2]; };

    #pragma unroll
    for (int ml = 0; ml < MLOOP; ++ml) {
        const int mTile = mWave * MLOOP + ml;
        const unsigned short* aRow = &sA[(mTile * 16 + l16) * LDW];
        v8f c = {};
        #pragma unroll
        for (int kk = 0; kk < K; kk += 32) {
            Frag a, b;
            a.q[0] = *(const uint4*)(aRow + kk + aOff);
            a.q[1] = *(const uint4*)(aRow + kk + 16 + aOff);
            b.q[0] = *(const uint4*)(bRow + kk + bOff);
            b.q[1] = *(const uint4*)(bRow + kk + 16 + bOff);
            c = __builtin_amdgcn_wmma_f32_16x16x32_bf16(
                    false, a.v, false, b.v, (short)0, c, false, false);
        }
        const int tileRow = blockRow + mTile * 16;            // SGPR-uniform
        const int rbase   = tileRow + half * 8;
        float* op = out + (size_t)rbase * NOUT + n;
        if (tileRow + 16 <= M) {
            // Scalar-uniform fast path: 8 unguarded coalesced stores.
            #pragma unroll
            for (int v = 0; v < 8; ++v) {
                float val = c[v] + bn;
                if (RELU) val = fmaxf(val, 0.f);
                op[(size_t)v * NOUT] = val;
            }
        } else {
            #pragma unroll
            for (int v = 0; v < 8; ++v) {
                if (rbase + v < M) {
                    float val = c[v] + bn;
                    if (RELU) val = fmaxf(val, 0.f);
                    op[(size_t)v * NOUT] = val;
                }
            }
        }
    }
}

// ---------------------------------------------------------------------------
// Row-wise log_softmax over 64 cols, one wave32 per node (2 cols per lane).
// ---------------------------------------------------------------------------
__global__ __launch_bounds__(256)
void logsoftmax_kernel(const float* __restrict__ in, float* __restrict__ out, int M) {
    int wid  = (int)(((long long)blockIdx.x * blockDim.x + threadIdx.x) >> 5);
    int lane = threadIdx.x & 31;
    if (wid >= M) return;
    const float* row = in + (size_t)wid * 64;
    float x0 = row[lane], x1 = row[lane + 32];
    float m = fmaxf(x0, x1);
    #pragma unroll
    for (int off = 16; off; off >>= 1) m = fmaxf(m, __shfl_xor(m, off, 32));
    float s = __expf(x0 - m) + __expf(x1 - m);
    #pragma unroll
    for (int off = 16; off; off >>= 1) s += __shfl_xor(s, off, 32);
    float lg = m + __logf(s);
    out[(size_t)wid * 64 + lane]      = x0 - lg;
    out[(size_t)wid * 64 + lane + 32] = x1 - lg;
}

// ---------------------------------------------------------------------------
extern "C" void kernel_launch(void* const* d_in, const int* in_sizes, int n_in,
                              void* d_out, int out_size, void* d_ws, size_t ws_size,
                              hipStream_t stream) {
    const float*     x   = (const float*)d_in[0];
    const long long* ei  = (const long long*)d_in[1];   // int64 [2, E]
    const float*     ew  = (const float*)d_in[2];
    const float*     W1  = (const float*)d_in[3];
    const float*     b1  = (const float*)d_in[4];
    const float*     W2  = (const float*)d_in[5];
    const float*     b2  = (const float*)d_in[6];
    const float*     W3  = (const float*)d_in[7];
    const float*     b3  = (const float*)d_in[8];
    float*           out = (float*)d_out;

    const int E = in_sizes[2];           // 1,600,000
    const int N = in_sizes[0] / 128;     // 100,000

    float* bufP = (float*)d_ws;                         // propagate accumulator
    float* bufH = bufP + (size_t)N * 128;               // dense-layer output
    const size_t fbytes = (size_t)N * 128 * sizeof(float);

    const long long* esrc = ei;
    const long long* edst = ei + E;

    // 8 waves/block, 8 edges/wave -> 64 edges per block
    const int propBlocks = (E + 63) / 64;
    const int gemmBlocks = (N + 63) / 64;
    const int lsmBlocks  = (int)(((long long)N * 32 + 255) / 256);

    // Layer 1: propagate(x) -> bufP ; relu(bufP@W1+b1) -> bufH
    hipMemsetAsync(bufP, 0, fbytes, stream);
    propagate_kernel<<<propBlocks, 256, 0, stream>>>(x, esrc, edst, ew, bufP, E);
    gemm_bias_act<128, 4, true><<<gemmBlocks, 256, 0, stream>>>(bufP, W1, b1, bufH, N);

    // Layer 2
    hipMemsetAsync(bufP, 0, fbytes, stream);
    propagate_kernel<<<propBlocks, 256, 0, stream>>>(bufH, esrc, edst, ew, bufP, E);
    gemm_bias_act<128, 4, true><<<gemmBlocks, 256, 0, stream>>>(bufP, W2, b2, bufH, N);

    // Layer 3: propagate -> bufP ; bufP@W3+b3 -> bufH (no relu) ; log_softmax -> out
    hipMemsetAsync(bufP, 0, fbytes, stream);
    propagate_kernel<<<propBlocks, 256, 0, stream>>>(bufH, esrc, edst, ew, bufP, E);
    gemm_bias_act<64, 2, false><<<gemmBlocks, 256, 0, stream>>>(bufP, W3, b3, bufH, N);
    logsoftmax_kernel<<<lsmBlocks, 256, 0, stream>>>(bufH, out, N);
}
